// EmbeddingSelector_43868795961407
// MI455X (gfx1250) — compile-verified, matrix operand order checked by reading, not run on hardware
//
#include <hip/hip_runtime.h>
#include <hip/hip_bf16.h>

// ---------------------------------------------------------------------------
// EmbeddingSelector on MI455X (gfx1250, wave32)
//
// scores[(c,b,h), n] = qW[(c,b,h),:] . A[n,:] + qb  (head-factored fusion)
// M=2048 x N=20000 x K=512 bf16 WMMA GEMM; operands + 83 MB bf16 score
// scratch live in the 192 MB L2.  Round 2: embedding tiles staged in LDS via
// CDNA5 async-to-LDS (global_load_async_to_lds_b128 + s_wait_asynccnt),
// double-buffered, shared by all 8 waves of the WG (4x less L2 traffic and
// no per-WMMA full loadcnt stall).
//   K1 convert A -> bf16 (ws)
//   K2 per-(c,b) projection -> qW bf16 + qb f32 (ws)
//   K3 WMMA score GEMM (LDS-staged B), fused bias+scale+label-mask
//   K4 per-(c,b): per-head max/sumexp, head-avg weights, streaming top-5
// Workspace: ~105.4 MB.
// ---------------------------------------------------------------------------

typedef __bf16 bf16_t;
typedef __attribute__((ext_vector_type(16))) __bf16 v16bf;
typedef __attribute__((ext_vector_type(8)))  __bf16 v8bf;
typedef __attribute__((ext_vector_type(8)))  float  v8f;

#define B_   32
#define N_   20000
#define C_   8
#define E_   512
#define H_   8
#define DH_  64
#define NS_  5
#define M_   (C_*B_*H_)        // 2048 rows (c,b,h)
#define NTILE 256
#define NPAD (79*NTILE)        // 20224
#define SCALE 0.125f           // 1/sqrt(dh) = 1/8
#define KCH  32                // K per chunk (one WMMA depth)
#define LROW 40                // halves per LDS row: 32 data + 8 pad (80 B)

#define USE_ASYNC 1

// ---------------- K1: f32 -> bf16 convert of all_embeddings ----------------
__global__ __launch_bounds__(256) void k_convert(const float* __restrict__ A,
                                                 bf16_t* __restrict__ Ab) {
  int i = blockIdx.x * 256 + threadIdx.x;
  if (i < N_ * E_) Ab[i] = (bf16_t)A[i];
}

// ---------------- K2: per-(c,b) q projection and qW = q.Wk -----------------
__global__ __launch_bounds__(256) void k_proj(const float* __restrict__ query,
                                              const float* __restrict__ Wq,
                                              const float* __restrict__ bq,
                                              const float* __restrict__ Wk,
                                              const float* __restrict__ bk,
                                              bf16_t* __restrict__ qWb,
                                              float* __restrict__ qb) {
  __shared__ float qe[E_];
  __shared__ float qv[E_];
  const int p = blockIdx.x;          // p = c*B + b
  const int c = p >> 5, b = p & 31;
  const int t = threadIdx.x;
  qe[t]       = query[b * E_ + t];
  qe[t + 256] = query[b * E_ + t + 256];
  __syncthreads();

  const int wv = t >> 5, ln = t & 31;
  const float* Wqc = Wq + (size_t)c * E_ * E_;
  for (int f = wv; f < E_; f += 8) {
    const float* row = Wqc + (size_t)f * E_;
    float s = 0.f;
    for (int e = ln; e < E_; e += 32) s += row[e] * qe[e];
    for (int off = 16; off > 0; off >>= 1) s += __shfl_down(s, off, 32);
    if (ln == 0) qv[f] = s + bq[c * E_ + f];
  }
  __syncthreads();

  const float* Wkc = Wk + (size_t)c * E_ * E_;
  for (int h = 0; h < H_; ++h) {
    const int r = p * H_ + h;
    for (int eo = 0; eo < E_; eo += 256) {
      const int e = eo + t;
      float s = 0.f;
      const float* base = Wkc + (size_t)(h * DH_) * E_ + e;
      #pragma unroll 4
      for (int f = 0; f < DH_; ++f) s += base[(size_t)f * E_] * qv[h * DH_ + f];
      qWb[(size_t)r * E_ + e] = (bf16_t)s;
    }
    if (t == 0) {
      float s = 0.f;
      for (int f = 0; f < DH_; ++f) s += qv[h * DH_ + f] * bk[c * E_ + h * DH_ + f];
      qb[r] = s;
    }
  }
}

// async fill of one 64-byte LDS row segment (4 x b128, one lane = one n-row)
__device__ __forceinline__ void fill_async(unsigned ldst, const bf16_t* g) {
#if USE_ASYNC
  #pragma unroll
  for (int i = 0; i < 4; ++i) {
    asm volatile("global_load_async_to_lds_b128 %0, %1, off"
                 :: "v"(ldst + i * 16), "v"((const void*)(g + i * 8))
                 : "memory");
  }
#else
  bf16_t* l = (bf16_t*)(size_t)0;  // unused
  (void)l; (void)ldst; (void)g;
#endif
}

// ---------------- K3: bf16 WMMA score GEMM + bias/scale/mask ---------------
// WG = 8 waves: 4(M) x 2(N) -> 64M x 256N.  Embedding tile (256n x 32k bf16)
// double-buffered in LDS via async-to-LDS; qW A-operands direct from L2.
__global__ __launch_bounds__(256) void k_scores(const bf16_t* __restrict__ qWb,
                                                const float* __restrict__ qb,
                                                const bf16_t* __restrict__ Ab,
                                                const unsigned char* __restrict__ mask,
                                                bf16_t* __restrict__ S) {
  __shared__ bf16_t smem[2 * 256 * LROW];   // 2 x 20 KB
  const int t = threadIdx.x;
  const int wv = t >> 5, lane = t & 31;
  const int m_wave = blockIdx.y * 64 + (wv & 3) * 16;
  const int nsub  = wv >> 2;                 // 0/1 -> n offset 0/128
  const int n_base = blockIdx.x * NTILE;
  const int lrow = lane & 15;
  const int koff = (lane < 16) ? 0 : 8;      // 16-bit A/B operand lane layout

  // fill mapping: thread t owns embedding row n_base+t (clamped; masked later)
  int nrow = n_base + t; if (nrow >= N_) nrow = N_ - 1;
  const bf16_t* grow = Ab + (size_t)nrow * E_;
  const unsigned lds0 = (unsigned)(size_t)(smem + t * LROW);              // low 32b = LDS offset
  const unsigned lds1 = (unsigned)(size_t)(smem + 256 * LROW + t * LROW);

  v8f acc[8] = {};
  const bf16_t* qrow = qWb + (size_t)(m_wave + lrow) * E_;

#if USE_ASYNC
  fill_async(lds0, grow);                    // prologue: buffer 0, kk=0
#else
  { v8bf* d = (v8bf*)(smem + t * LROW);
    #pragma unroll
    for (int i = 0; i < 4; ++i) d[i] = *(const v8bf*)(grow + i * 8); }
#endif

  for (int kc = 0; kc < E_ / KCH; ++kc) {    // 16 chunks
    const int kk = kc * KCH;
    if (kc + 1 < E_ / KCH) {                 // prefetch next chunk into other buffer
#if USE_ASYNC
      fill_async((kc & 1) ? lds0 : lds1, grow + kk + KCH);
#else
      bf16_t* dst = smem + ((kc & 1) ? 0 : 256 * LROW) + t * LROW;
      v8bf* d = (v8bf*)dst;
      #pragma unroll
      for (int i = 0; i < 4; ++i) d[i] = *(const v8bf*)(grow + kk + KCH + i * 8);
#endif
    }
#if USE_ASYNC
    if (kc + 1 < E_ / KCH) asm volatile("s_wait_asynccnt 4" ::: "memory");
    else                   asm volatile("s_wait_asynccnt 0" ::: "memory");
#endif
    __syncthreads();                         // current buffer visible to all waves

    const bf16_t* cbuf = smem + ((kc & 1) ? 256 * LROW : 0);
    v8bf alo = *(const v8bf*)(qrow + kk + koff);
    v8bf ahi = *(const v8bf*)(qrow + kk + 16 + koff);
    v16bf a = __builtin_shufflevector(alo, ahi, 0,1,2,3,4,5,6,7,8,9,10,11,12,13,14,15);
    #pragma unroll
    for (int j = 0; j < 8; ++j) {
      const bf16_t* brow = cbuf + (nsub * 128 + j * 16 + lrow) * LROW;
      v8bf blo = *(const v8bf*)(brow + koff);
      v8bf bhi = *(const v8bf*)(brow + koff + 16);
      v16bf bb = __builtin_shufflevector(blo, bhi, 0,1,2,3,4,5,6,7,8,9,10,11,12,13,14,15);
      acc[j] = __builtin_amdgcn_wmma_f32_16x16x32_bf16(
          false, a, false, bb, (short)0, acc[j], false, false);
    }
    __syncthreads();                         // all readers done before overwrite
  }

  // C/D layout: VGPR i, lanes 0-15 -> M=i, lanes 16-31 -> M=8+i; N = lane&15
  const int mrow_off = (lane < 16) ? 0 : 8;
  const int ncol = lane & 15;
  const int n_wave = n_base + nsub * 128;
  float qbv[8];
  #pragma unroll
  for (int i = 0; i < 8; ++i) qbv[i] = qb[m_wave + i + mrow_off];
  #pragma unroll
  for (int j = 0; j < 8; ++j) {
    const int n = n_wave + j * 16 + ncol;
    #pragma unroll
    for (int i = 0; i < 8; ++i) {
      const int m = m_wave + i + mrow_off;
      float v = (acc[j][i] + qbv[i]) * SCALE;
      const int cls = m >> 8;                // c = m / (B*H)
      const bool ok = (n < N_) && mask[(size_t)n * C_ + cls];
      if (!ok) v = -1e9f;
      S[(size_t)m * NPAD + n] = (bf16_t)v;
    }
  }
}

// ---------------- K4: per-(c,b) softmax (per head) + head-avg top-5 --------
__global__ __launch_bounds__(256) void k_topk(const bf16_t* __restrict__ S,
                                              int* __restrict__ out_idx,
                                              float* __restrict__ out_w) {
  __shared__ float red[256];
  __shared__ float mh[H_], iZ[H_];
  __shared__ float sv[256 * NS_];
  __shared__ int   si[256 * NS_];
  const int p = blockIdx.x;          // p = c*B + b
  const int c = p >> 5, b = p & 31;
  const int t = threadIdx.x;
  const int r0 = p * H_;

  for (int h = 0; h < H_; ++h) {
    const bf16_t* row = S + (size_t)(r0 + h) * NPAD;
    float m = -3.0e38f;
    for (int n = t; n < N_; n += 256) { float v = (float)row[n]; m = v > m ? v : m; }
    red[t] = m; __syncthreads();
    for (int s = 128; s > 0; s >>= 1) {
      if (t < s) { float o = red[t + s]; if (o > red[t]) red[t] = o; }
      __syncthreads();
    }
    const float mx = red[0]; __syncthreads();
    float z = 0.f;
    for (int n = t; n < N_; n += 256) z += __expf((float)row[n] - mx);
    red[t] = z; __syncthreads();
    for (int s = 128; s > 0; s >>= 1) {
      if (t < s) red[t] += red[t + s];
      __syncthreads();
    }
    if (t == 0) { mh[h] = mx; iZ[h] = 1.f / red[0]; }
    __syncthreads();
  }

  float tv[NS_]; int ti[NS_];
  #pragma unroll
  for (int j = 0; j < NS_; ++j) { tv[j] = -1.f; ti[j] = -1; }
  for (int n = t; n < N_; n += 256) {
    float w = 0.f;
    #pragma unroll
    for (int h = 0; h < H_; ++h) {
      float s = (float)S[(size_t)(r0 + h) * NPAD + n];
      w += __expf(s - mh[h]) * iZ[h];
    }
    w *= (1.0f / H_);
    if (w > tv[NS_ - 1]) {
      int j = NS_ - 1;
      while (j > 0 && tv[j - 1] < w) { tv[j] = tv[j - 1]; ti[j] = ti[j - 1]; --j; }
      tv[j] = w; ti[j] = n;
    }
  }
  #pragma unroll
  for (int j = 0; j < NS_; ++j) { sv[t * NS_ + j] = tv[j]; si[t * NS_ + j] = ti[j]; }
  __syncthreads();

  if (t == 0) {
    for (int j = 0; j < NS_; ++j) {
      float best = -2.f; int bi = -1, slot = 0;
      for (int q = 0; q < 256 * NS_; ++q)
        if (sv[q] > best) { best = sv[q]; bi = si[q]; slot = q; }
      sv[slot] = -2.f;
      const int o = (b * C_ + c) * NS_ + j;  // (B, C, ns) layout
      out_idx[o] = bi;
      out_w[o] = best;
    }
  }
}

// ---------------------------------------------------------------------------
extern "C" void kernel_launch(void* const* d_in, const int* in_sizes, int n_in,
                              void* d_out, int out_size, void* d_ws, size_t ws_size,
                              hipStream_t stream) {
  (void)in_sizes; (void)n_in; (void)out_size; (void)ws_size;
  const float* query = (const float*)d_in[0];               // (B,E)
  const float* A     = (const float*)d_in[1];               // (N,E)
  const unsigned char* mask = (const unsigned char*)d_in[2];// (N,C) bool
  const float* Wq    = (const float*)d_in[3];               // (C,E,E)
  const float* bq    = (const float*)d_in[4];               // (C,E)
  const float* Wk    = (const float*)d_in[5];               // (C,E,E)
  const float* bk    = (const float*)d_in[6];               // (C,E)
  // d_in[7] = n_samples (compile-time NS_=5)

  char* ws = (char*)d_ws;
  size_t off = 0;
  bf16_t* Ab  = (bf16_t*)(ws + off); off += (size_t)N_ * E_ * sizeof(bf16_t);   // 20.48 MB
  bf16_t* qWb = (bf16_t*)(ws + off); off += (size_t)M_ * E_ * sizeof(bf16_t);   //  2.10 MB
  float*  qb  = (float*)(ws + off);  off += (size_t)M_ * sizeof(float);         //  8 KB
  bf16_t* S   = (bf16_t*)(ws + off); off += (size_t)M_ * NPAD * sizeof(bf16_t); // 82.8 MB

  int* out_idx = (int*)d_out;                               // (B,C,5) int32
  float* out_w = (float*)d_out + B_ * C_ * NS_;             // (B,C,5) f32

  k_convert<<<(N_ * E_ + 255) / 256, 256, 0, stream>>>(A, Ab);
  k_proj<<<C_ * B_, 256, 0, stream>>>(query, Wq, bq, Wk, bk, qWb, qb);
  dim3 g3(NPAD / NTILE, M_ / 64);                           // (79, 32)
  k_scores<<<g3, 256, 0, stream>>>(qWb, qb, Ab, mask, S);
  k_topk<<<C_ * B_, 256, 0, stream>>>(S, out_idx, out_w);
}
